// MyModel_83854941487657
// MI455X (gfx1250) — matrix operand order repaired and checked by reading, not compiled, not run on hardware
//
#include <hip/hip_runtime.h>
#include <cstdint>

typedef __attribute__((ext_vector_type(16))) _Float16 v16h;
typedef __attribute__((ext_vector_type(8)))  _Float16 v8h;
typedef __attribute__((ext_vector_type(8)))  float    v8f;

#define NNODES 16384
#define EMAX   540672
#define EPSV   1e-5f

// ---- float <-> order-preserving u32 (for atomicMax-based segment_max) ----
__device__ __forceinline__ unsigned fenc(float f) {
  unsigned u = __float_as_uint(f);
  return (u & 0x80000000u) ? ~u : (u | 0x80000000u);
}
__device__ __forceinline__ float fdec(unsigned u) {
  return __uint_as_float((u & 0x80000000u) ? (u ^ 0x80000000u) : ~u);
}
// fenc(-inf) == 0x007FFFFF  (init value for max buffers)

// ---------------- fragment-layout index helpers (CDNA5 WMMA) -------------
// A-matrix f16 16x32 fragment: lanes 0-15 hold K {0..7,16..23} (e=0..15),
// lanes 16-31 hold K {8..15,24..31}. Storage: frag-major, lane-major, e.
__device__ __forceinline__ size_t aidx(int row, int k, int KP) {
  int rt = row >> 4, m = row & 15;
  int kc = k >> 5, kin = k & 31;
  int lane = m + (((kin >> 3) & 1) << 4);
  int kk = kin & ~8;                 // {0..7} or {16..23}
  int e  = (kk < 8) ? kk : kk - 8;
  return ((size_t)(rt * (KP >> 5) + kc) * 32 + lane) * 16 + e;
}
// B-matrix f16 32x16 fragment: lanes 0-15 K {0..7,16..23}, lanes 16-31 K {8..15,24..31}.
__device__ __forceinline__ size_t widx(int k, int n, int NOUT) {
  int kc = k >> 5, nt = n >> 4;
  int kin = k & 31;
  int lane = (n & 15) + (((kin >> 3) & 1) << 4);
  int kk = kin & ~8;
  int e = (kk < 8) ? kk : kk - 8;
  return ((size_t)(kc * (NOUT >> 4) + nt) * 32 + lane) * 16 + e;
}
// C/D f32 16x16 tile: VGPR r, lanes 0-15 M=r, lanes 16-31 M=8+r; N=lane%16.
__device__ __forceinline__ size_t yidx(int row, int col, int NOUT) {
  int rt = row >> 4, nt = col >> 4;
  int lane = (col & 15) + ((row & 8) ? 16 : 0);
  int r = row & 7;
  return ((size_t)(rt * (NOUT >> 4) + nt) * 32 + lane) * 8 + r;
}

// ------------------------- small utility kernels -------------------------
__global__ void k_fill_u32(unsigned* __restrict__ p, unsigned v, int n) {
  int i = blockIdx.x * blockDim.x + threadIdx.x;
  if (i < n) p[i] = v;
}
__global__ void k_zero_f32(float* __restrict__ p, int n) {
  int i = blockIdx.x * blockDim.x + threadIdx.x;
  if (i < n) p[i] = 0.f;
}
// weights f32[fin,fout] -> f16 B-fragment layout [KP,fout], zero-padded K
__global__ void k_convw(const float* __restrict__ W, _Float16* __restrict__ Wh,
                        int fin, int fout, int KP) {
  int i = blockIdx.x * blockDim.x + threadIdx.x;
  if (i >= KP * fout) return;
  int k = i / fout, n = i - k * fout;
  float v = (k < fin) ? W[i] : 0.f;
  Wh[widx(k, n, fout)] = (_Float16)v;
}

// ------------------------------ gathers ---------------------------------
// conv1 edge input: [E,32] = [x[src](3) | pos[src]-pos[min(dst,N-1)](3) | 0]
__global__ void k_gather1(const float* __restrict__ x, const float* __restrict__ pos,
                          const int* __restrict__ src, const int* __restrict__ dst,
                          _Float16* __restrict__ A, int total) {
  int i = blockIdx.x * blockDim.x + threadIdx.x;
  if (i >= total) return;
  int e = i >> 5, k = i & 31;
  float v = 0.f;
  if (k < 6) {
    int s = src[e];
    if (k < 3) {
      v = x[s * 3 + k];
    } else {
      int d = dst[e]; if (d > NNODES - 1) d = NNODES - 1;
      int j = k - 3;
      v = pos[s * 3 + j] - pos[d * 3 + j];
    }
  }
  A[aidx(e, k, 32)] = (_Float16)v;
}
// conv2 edge input: [E,160] = [h1[src](128) | rel(3) | 0...]
__global__ void k_gather2(const unsigned* __restrict__ h1, const float* __restrict__ pos,
                          const int* __restrict__ src, const int* __restrict__ dst,
                          _Float16* __restrict__ A, int total) {
  int i = blockIdx.x * blockDim.x + threadIdx.x;
  if (i >= total) return;
  int e = i / 160, k = i - e * 160;
  float v = 0.f;
  if (k < 128) {
    v = fdec(h1[(size_t)src[e] * 128 + k]);
  } else if (k < 131) {
    int s = src[e];
    int d = dst[e]; if (d > NNODES - 1) d = NNODES - 1;
    int j = k - 128;
    v = pos[s * 3 + j] - pos[d * 3 + j];
  }
  A[aidx(e, k, 160)] = (_Float16)v;
}
// pool input: [N,288] = [h2(256) | pos(3) | 0...]
__global__ void k_gather3(const unsigned* __restrict__ h2, const float* __restrict__ pos,
                          _Float16* __restrict__ A, int total) {
  int i = blockIdx.x * blockDim.x + threadIdx.x;
  if (i >= total) return;
  int node = i / 288, k = i - node * 288;
  float v = 0.f;
  if (k < 256)      v = fdec(h2[(size_t)node * 256 + k]);
  else if (k < 259) v = pos[node * 3 + (k - 256)];
  A[aidx(node, k, 288)] = (_Float16)v;
}
// encoded graph max -> f16 fc input [16,1024] (fragment layout)
__global__ void k_g2f(const unsigned* __restrict__ g, _Float16* __restrict__ A, int n) {
  int i = blockIdx.x * blockDim.x + threadIdx.x;
  if (i >= n) return;
  int row = i >> 10, k = i & 1023;
  A[aidx(row, k, 1024)] = (_Float16)fdec(g[i]);
}

// -------------------- WMMA GEMM + bias + ReLU + stats --------------------
// All operands in fragment layout; no LDS, no barriers. Block = 256 thr =
// 8 waves (4 row-tiles x 2 col-groups); each wave owns 16 x (TN*16) output.
template<int TN>
__global__ __launch_bounds__(256) void k_gemm_frag(
    const _Float16* __restrict__ A, const _Float16* __restrict__ Wf,
    const float* __restrict__ bias, _Float16* __restrict__ Y,
    float* __restrict__ stats, int rows, int KP, int NOUT)
{
  const int lane = threadIdx.x & 31;
  const int wid  = threadIdx.x >> 5;
  const int wm   = wid & 3;
  const int wn   = wid >> 2;
  const int rt   = blockIdx.x * 4 + wm;            // 16-row tile index
  const int ct0  = blockIdx.y * (2 * TN) + wn * TN; // 16-col tile index
  const int KC   = KP >> 5;
  const int NT   = NOUT >> 4;

  v8f acc[TN] = {};

  const _Float16* Ap = A  + ((size_t)rt  * KC * 32 + lane) * 16;
  const _Float16* Wp = Wf + ((size_t)ct0 * 32       + lane) * 16;
  const size_t wstride = (size_t)NT * 512;          // per-kc W advance (halfs)

  for (int kc = 0; kc < KC; ++kc) {
    const _Float16* ap = Ap + (size_t)kc * 512;
    v8h a0 = *(const v8h*)ap;
    v8h a1 = *(const v8h*)(ap + 8);
    __builtin_prefetch(ap + 512, 0, 0);            // speculative next chunk
    v16h af = __builtin_shufflevector(a0, a1,
        0,1,2,3,4,5,6,7,8,9,10,11,12,13,14,15);
    const _Float16* wp = Wp + (size_t)kc * wstride;
#pragma unroll
    for (int j = 0; j < TN; ++j) {
      v8h b0 = *(const v8h*)(wp + (size_t)j * 512);
      v8h b1 = *(const v8h*)(wp + (size_t)j * 512 + 8);
      v16h bf = __builtin_shufflevector(b0, b1,
          0,1,2,3,4,5,6,7,8,9,10,11,12,13,14,15);
      acc[j] = __builtin_amdgcn_wmma_f32_16x16x32_f16(
          false, af, false, bf, (short)0, acc[j], false, false);
    }
  }

  // epilogue: bias + ReLU, b128 tile store, column sum/sumsq -> atomics
  if (rt * 16 < rows) {
    const int n = lane & 15;
#pragma unroll
    for (int j = 0; j < TN; ++j) {
      const int gcol = (ct0 + j) * 16 + n;
      const float bi = bias[gcol];
      float s = 0.f, s2 = 0.f;
      v8h yv;
#pragma unroll
      for (int r = 0; r < 8; ++r) {
        float y = acc[j][r] + bi;
        y = y > 0.f ? y : 0.f;
        yv[r] = (_Float16)y;
        s += y; s2 += y * y;
      }
      *(v8h*)(Y + ((size_t)(rt * NT + ct0 + j) * 32 + lane) * 8) = yv;
      s  += __shfl_xor(s, 16);
      s2 += __shfl_xor(s2, 16);
      if (lane < 16) {
        atomicAdd(stats + gcol, s);
        atomicAdd(stats + NOUT + gcol, s2);
      }
    }
  }
}

// normalize (batchnorm over rows); Y (D-frag layout) -> A (A-frag layout)
__global__ void k_norm_cast(const _Float16* __restrict__ Y, const float* __restrict__ stats,
                            const float* __restrict__ gamma, const float* __restrict__ beta,
                            _Float16* __restrict__ Aout, int total, int NOUT, float invR) {
  int i = blockIdx.x * blockDim.x + threadIdx.x;
  if (i >= total) return;
  int row = i / NOUT, col = i - row * NOUT;
  float mu  = stats[col] * invR;
  float var = stats[NOUT + col] * invR - mu * mu;
  float v = gamma[col] * ((float)Y[yidx(row, col, NOUT)] - mu) *
            rsqrtf(var + EPSV) + beta[col];
  Aout[aidx(row, col, NOUT)] = (_Float16)v;
}

// segment_max over edges (msg in A-frag layout) -> encoded node buffer
__global__ void k_segmax_edge(const _Float16* __restrict__ msg, const int* __restrict__ dst,
                              unsigned* __restrict__ nodes, int total, int C) {
  int i = blockIdx.x * blockDim.x + threadIdx.x;
  if (i >= total) return;
  int e = i / C, c = i - e * C;
  int d = dst[e];
  if (d < NNODES) atomicMax(nodes + (size_t)d * C + c, fenc((float)msg[aidx(e, c, C)]));
}
// segment_max over batch (h in A-frag layout) -> encoded graph buffer
__global__ void k_segmax_batch(const _Float16* __restrict__ h, const int* __restrict__ batch,
                               unsigned* __restrict__ g, int total, int C) {
  int i = blockIdx.x * blockDim.x + threadIdx.x;
  if (i >= total) return;
  int node = i / C, c = i - node * C;
  atomicMax(g + (size_t)batch[node] * C + c, fenc((float)h[aidx(node, c, C)]));
}

// final Linear 256 -> 1 over 16 rows (A in A-frag layout)
__global__ void k_fc_out(const _Float16* __restrict__ A, const float* __restrict__ W3,
                         const float* __restrict__ b3, float* __restrict__ out) {
  int i = threadIdx.x;
  if (i < 16) {
    float s = 0.f;
    for (int k = 0; k < 256; ++k) s += (float)A[aidx(i, k, 256)] * W3[k];
    out[i] = s + b3[0];
  }
}

// ------------------------------- driver ----------------------------------
extern "C" void kernel_launch(void* const* d_in, const int* in_sizes, int n_in,
                              void* d_out, int out_size, void* d_ws, size_t ws_size,
                              hipStream_t stream) {
  (void)in_sizes; (void)n_in; (void)out_size;

  const float* x   = (const float*)d_in[0];
  const float* pos = (const float*)d_in[1];
  const int* src   = (const int*)d_in[2];
  const int* dst   = (const int*)d_in[3];
  const int* batch = (const int*)d_in[4];

  char* ws = (char*)d_ws;
  size_t oA  = 0;                                   // f16 activations [E,160] max
  size_t oY  = oA  + (size_t)EMAX * 160 * 2;        // f16 pre-norm Y  [E,256] max
  size_t oN1 = oY  + (size_t)EMAX * 256 * 2;        // enc u32 [N,128]
  size_t oN2 = oN1 + (size_t)NNODES * 128 * 4;      // enc u32 [N,256]
  size_t oG  = oN2 + (size_t)NNODES * 256 * 4;      // enc u32 [16,1024]
  size_t oS  = oG  + (size_t)16 * 1024 * 4;         // stats f32 [2048]
  size_t oW  = oS  + (size_t)2048 * 4;              // f16 weight scratch (2 MB)
  size_t need = oW + (size_t)1024 * 1024 * 2;
  if (ws_size < need) return;

  _Float16* bufA = (_Float16*)(ws + oA);
  _Float16* bufY = (_Float16*)(ws + oY);
  unsigned* n1   = (unsigned*)(ws + oN1);
  unsigned* n2   = (unsigned*)(ws + oN2);
  unsigned* g    = (unsigned*)(ws + oG);
  float*    st   = (float*)(ws + oS);
  _Float16* Wbuf = (_Float16*)(ws + oW);

  auto cdiv = [](int a, int b) { return (a + b - 1) / b; };

  // one MLP layer: convert weights, zero stats, GEMM+ReLU+stats, normalize
  auto layer = [&](int rows, int KP, int fin, int fout, int wi) {
    k_convw<<<cdiv(KP * fout, 256), 256, 0, stream>>>(
        (const float*)d_in[wi], Wbuf, fin, fout, KP);
    k_zero_f32<<<cdiv(2 * fout, 256), 256, 0, stream>>>(st, 2 * fout);
    const float* bias = (const float*)d_in[wi + 1];
    if (fout == 64) {
      dim3 gg(cdiv(rows, 64), 1);
      k_gemm_frag<2><<<gg, 256, 0, stream>>>(bufA, Wbuf, bias, bufY, st, rows, KP, fout);
    } else if (fout == 128) {
      dim3 gg(cdiv(rows, 64), 1);
      k_gemm_frag<4><<<gg, 256, 0, stream>>>(bufA, Wbuf, bias, bufY, st, rows, KP, fout);
    } else {                                   // fout multiple of 256
      dim3 gg(cdiv(rows, 64), fout / 256);
      k_gemm_frag<8><<<gg, 256, 0, stream>>>(bufA, Wbuf, bias, bufY, st, rows, KP, fout);
    }
    k_norm_cast<<<cdiv(rows * fout, 256), 256, 0, stream>>>(
        bufY, st, (const float*)d_in[wi + 2], (const float*)d_in[wi + 3],
        bufA, rows * fout, fout, 1.0f / (float)rows);
  };

  const unsigned NEG_INF_ENC = 0x007FFFFFu;

  // ---- conv1: [6->64->64->128] over edges ----
  k_gather1<<<cdiv(EMAX * 32, 256), 256, 0, stream>>>(x, pos, src, dst, bufA, EMAX * 32);
  layer(EMAX, 32, 6,  64,  5);
  layer(EMAX, 64, 64, 64,  9);
  layer(EMAX, 64, 64, 128, 13);
  k_fill_u32<<<cdiv(NNODES * 128, 256), 256, 0, stream>>>(n1, NEG_INF_ENC, NNODES * 128);
  k_segmax_edge<<<cdiv(EMAX * 128, 256), 256, 0, stream>>>(bufA, dst, n1, EMAX * 128, 128);

  // ---- conv2: [131->128->128->256] over edges ----
  k_gather2<<<cdiv(EMAX * 160, 256), 256, 0, stream>>>(n1, pos, src, dst, bufA, EMAX * 160);
  layer(EMAX, 160, 131, 128, 17);
  layer(EMAX, 128, 128, 128, 21);
  layer(EMAX, 128, 128, 256, 25);
  k_fill_u32<<<cdiv(NNODES * 256, 256), 256, 0, stream>>>(n2, NEG_INF_ENC, NNODES * 256);
  k_segmax_edge<<<cdiv(EMAX * 256, 256), 256, 0, stream>>>(bufA, dst, n2, EMAX * 256, 256);

  // ---- pool MLP: [259->256->512->1024] over nodes ----
  k_gather3<<<cdiv(NNODES * 288, 256), 256, 0, stream>>>(n2, pos, bufA, NNODES * 288);
  layer(NNODES, 288, 259, 256,  29);
  layer(NNODES, 256, 256, 512,  33);
  layer(NNODES, 512, 512, 1024, 37);
  k_fill_u32<<<cdiv(16 * 1024, 256), 256, 0, stream>>>(g, NEG_INF_ENC, 16 * 1024);
  k_segmax_batch<<<cdiv(NNODES * 1024, 256), 256, 0, stream>>>(bufA, batch, g,
                                                               NNODES * 1024, 1024);

  // ---- fc: [1024->512->256] + Linear(256->1) over 16 graphs ----
  k_g2f<<<cdiv(16 * 1024, 256), 256, 0, stream>>>(g, bufA, 16 * 1024);
  layer(16, 1024, 1024, 512, 41);
  layer(16, 512,  512,  256, 45);
  k_fc_out<<<1, 64, 0, stream>>>(bufA, (const float*)d_in[49],
                                 (const float*)d_in[50], (float*)d_out);
}
